// EQGATGNN_46394236731783
// MI455X (gfx1250) — matrix operand order repaired and checked by reading, not compiled, not run on hardware
//
#include <hip/hip_runtime.h>
#include <hip/hip_bf16.h>

// ---------------------------------------------------------------------------
// EQGAT-GNN forward for MI455X (gfx1250, wave32, WMMA).
//
// Node-sized tensors (4-48MB) are L2-resident (192MB L2), so the edge pipeline
// (edge_net1 -> SiLU -> edge_net2 -> cutoff -> message scatter) is fused into
// one kernel per layer: a wave owns 16 edges, gathers the [16 x 288] A-operand
// (s[i] | s[j] | bessel) from L2-resident f16 buffers, runs the 288->128 GEMM
// with v_wmma_f32_16x16x32_f16, stages the SiLU'd tile in LDS, runs the
// 128->{256|512} GEMM chunk-wise and scatters messages with f32 L2 atomics.
//
// Inner-loop scheduling: all B-fragment loads of a k-step are issued as one
// clause, then a sched_barrier(0) pins them before the 8 back-to-back WMMAs so
// the post-RA scheduler cannot recreate the load->wait->wmma ping-pong (one
// s_wait_loadcnt per k-step instead of eight).
//
// WMMA fragment layouts as used here (C/D as in CDNA5 ISA 7.12.2):
//   A (16x32 f16): row m = lane&15; halves k = K0+(lane>>4)*16..+15 (contiguous)
//   B: weights pre-transposed to [N,K] f16; lane holds column n = lane&15,
//      K-half chosen by lane>>4 (16 contiguous halves).
//   C/D (16x16 f32, 8 VGPR): M = vgpr + (lane>>4)*8, N = lane&15.
// ---------------------------------------------------------------------------

typedef __attribute__((ext_vector_type(16))) _Float16 v16h;
typedef __attribute__((ext_vector_type(8)))  _Float16 v8h;
typedef __attribute__((ext_vector_type(8)))  float    v8f;

#define FLAG_SILU  1
#define FLAG_ACCUM 2

#if defined(__has_builtin)
#if __has_builtin(__builtin_amdgcn_sched_barrier)
#define SCHED_FENCE() __builtin_amdgcn_sched_barrier(0)
#endif
#endif
#ifndef SCHED_FENCE
#define SCHED_FENCE() ((void)0)
#endif

static __device__ __forceinline__ v16h ldfrag(const _Float16* p0, const _Float16* p1) {
  v8h a0 = *reinterpret_cast<const v8h*>(p0);
  v8h a1 = *reinterpret_cast<const v8h*>(p1);
  v16h r;
#pragma unroll
  for (int q = 0; q < 8; ++q) { r[q] = a0[q]; r[8 + q] = a1[q]; }
  return r;
}

static __device__ __forceinline__ v8f wmma_f16(v16h a, v16h b, v8f c) {
  return __builtin_amdgcn_wmma_f32_16x16x32_f16(false, a, false, b, (short)0, c, false, false);
}

// ---------------------------------------------------------------- utilities
__global__ void k_zero(float* __restrict__ p, size_t n) {
  size_t i = (size_t)blockIdx.x * blockDim.x + threadIdx.x;
  if (i < n) p[i] = 0.f;
}

__global__ void k_f2h(const float* __restrict__ s, _Float16* __restrict__ d, size_t n) {
  size_t i = (size_t)blockIdx.x * blockDim.x + threadIdx.x;
  if (i < n) d[i] = (_Float16)s[i];
}

// src: f32 [K,N] row-major  ->  dst: f16 [N,K] row-major (i.e. transposed)
__global__ void k_transT(const float* __restrict__ src, _Float16* __restrict__ dst,
                         int K, int N) {
  int i = blockIdx.x * blockDim.x + threadIdx.x;
  if (i >= K * N) return;
  int k = i / N, n = i % N;
  dst[(size_t)n * K + k] = (_Float16)src[(size_t)k * N + n];
}

// -------------------------------------------------------------- edge geometry
__global__ void k_geom(const float* __restrict__ pos,
                       const int* __restrict__ ej, const int* __restrict__ ei,
                       float* __restrict__ dArr, float* __restrict__ cutE,
                       float* __restrict__ rvec, _Float16* __restrict__ deh,
                       float* __restrict__ cnt, int E) {
  int e = blockIdx.x * blockDim.x + threadIdx.x;
  if (e >= E) return;
  int jj = ej[e], ii = ei[e];
  float px = pos[3 * jj + 0] - pos[3 * ii + 0];
  float py = pos[3 * jj + 1] - pos[3 * ii + 1];
  float pz = pos[3 * jj + 2] - pos[3 * ii + 2];
  float d = sqrtf(px * px + py * py + pz * pz);
  float inv = 1.f / d;
  dArr[e] = d;
  rvec[3 * e + 0] = px * inv;
  rvec[3 * e + 1] = py * inv;
  rvec[3 * e + 2] = pz * inv;
  const float PI5 = 0.62831853071795864769f;       // pi / CUTOFF
  float cut = (d < 5.f) ? 0.5f * (__cosf(d * PI5) + 1.f) : 0.f;
  cutE[e] = cut;
  const float c0 = 0.632455532033675866f;           // sqrt(2/CUTOFF)
#pragma unroll
  for (int n = 1; n <= 32; ++n)
    deh[(size_t)e * 32 + (n - 1)] = (_Float16)(c0 * __sinf((float)n * PI5 * d) * inv);
  atomicAdd(&cnt[ii], 1.f);
}

// -------------------------------- neighbor embedding (2 edges per 256-thread block)
__global__ __launch_bounds__(256) void k_nbr(
    const float* __restrict__ dArr, const float* __restrict__ cutE,
    const int* __restrict__ z, const int* __restrict__ ej, const int* __restrict__ ei,
    const float* __restrict__ dpW, const float* __restrict__ dpB,
    const float* __restrict__ nemb, float* __restrict__ agg) {
  __shared__ float rb[2][32];
  int sub = threadIdx.x >> 7;                 // which edge in the block
  int t = threadIdx.x & 127;                  // feature index
  int e = blockIdx.x * 2 + sub;
  float d = dArr[e], cut = cutE[e];
  if (t < 32) {
    const float start = 0.00673794699908546710f;    // exp(-CUTOFF)
    float mean = start + (1.f - start) * (float)t * (1.f / 31.f);
    float tmp = (2.f / 32.f) * (1.f - start);
    float beta = 1.f / (tmp * tmp);
    float u = __expf(-d) - mean;                     // alpha = 5/CUTOFF = 1
    rb[sub][t] = cut * __expf(-beta * u * u);
  }
  __syncthreads();
  float acc = 0.f;
#pragma unroll
  for (int k = 0; k < 32; ++k) acc += rb[sub][k] * dpW[k * 128 + t];
  float Wv = (acc + dpB[t]) * cut;
  int jj = ej[e], ii = ei[e];
  atomicAdd(&agg[(size_t)ii * 128 + t], nemb[(size_t)z[jj] * 128 + t] * Wv);
}

__global__ void k_cat0(const int* __restrict__ z, const float* __restrict__ emb,
                       const float* __restrict__ agg, _Float16* __restrict__ cat, int N) {
  int i = blockIdx.x * blockDim.x + threadIdx.x;
  if (i >= N * 128) return;
  int n = i >> 7, h = i & 127;
  cat[(size_t)n * 256 + h]       = (_Float16)emb[(size_t)z[n] * 128 + h];
  cat[(size_t)n * 256 + 128 + h] = (_Float16)agg[i];
}

// -------------------------------------------------------- generic WMMA GEMM
// C[M,NTOT] = A[M,K](f16) @ Wt^T (Wt is f16 [NTOT,K]) + bias; optional SiLU,
// optional f16 copy of the output, optional in-place accumulate into C.
// Wave computes a 16x128 strip. block = 128 threads = 4 waves = 64 rows.
// Per k-step: A clause + 8-tile B clause -> sched fence -> 8 WMMAs.
__global__ __launch_bounds__(128) void k_gemm(
    const _Float16* __restrict__ A, const _Float16* __restrict__ Wt, int K,
    const float* __restrict__ bias, float* __restrict__ C, _Float16* __restrict__ Ch,
    int NTOT, int flags) {
  const int lane = threadIdx.x & 31, w = threadIdx.x >> 5;
  const int row = lane & 15, half = lane >> 4;
  const int m0 = (blockIdx.x * 4 + w) * 16;
  const int n0 = blockIdx.y * 128;
  const _Float16* arow = A + (size_t)(m0 + row) * K;
  const _Float16* wrow = Wt + (size_t)(n0 + row) * K + half * 16;

  v8f acc[8];
#pragma unroll
  for (int t = 0; t < 8; ++t)
#pragma unroll
    for (int q = 0; q < 8; ++q) acc[t][q] = 0.f;

  for (int k0 = 0; k0 < K; k0 += 32) {
    v16h a = ldfrag(arow + k0 + half * 16, arow + k0 + half * 16 + 8);
    __builtin_prefetch(arow + k0 + 32, 0, 3);       // global_prefetch_b8 (WGP scope)
    v16h bf[8];
#pragma unroll
    for (int t = 0; t < 8; ++t) {                   // issue all B loads first
      const _Float16* wr = wrow + (size_t)t * 16 * K + k0;
      bf[t] = ldfrag(wr, wr + 8);
    }
    SCHED_FENCE();                                  // loads stay before WMMAs
#pragma unroll
    for (int t = 0; t < 8; ++t)                     // 8 back-to-back WMMAs
      acc[t] = wmma_f16(a, bf[t], acc[t]);
    SCHED_FENCE();
  }
#pragma unroll
  for (int t = 0; t < 8; ++t) {
    int n = n0 + t * 16 + row;
    float bb = bias ? bias[n] : 0.f;
#pragma unroll
    for (int v = 0; v < 8; ++v) {
      float x = acc[t][v] + bb;
      if (flags & FLAG_SILU) x = x * (1.f / (1.f + __expf(-x)));
      size_t o = (size_t)(m0 + v + half * 8) * NTOT + n;
      if (C) { if (flags & FLAG_ACCUM) C[o] += x; else C[o] = x; }
      if (Ch) Ch[o] = (_Float16)x;
    }
  }
}

// ----------------------------------------------- fused EQGAT edge convolution
static __device__ __forceinline__ const _Float16*
asrc(const _Float16* si, const _Float16* sj, const _Float16* de, int k) {
  if (k < 128) return si + k;
  if (k < 256) return sj + (k - 128);
  return de + (k - 256);
}

__global__ __launch_bounds__(128) void k_edge_conv(
    const _Float16* __restrict__ sf16, const _Float16* __restrict__ deh,
    const float* __restrict__ cutE, const float* __restrict__ rvec,
    const float* __restrict__ sb, const float* __restrict__ vcur,
    const float* __restrict__ vb,
    const int* __restrict__ ej, const int* __restrict__ ei,
    const _Float16* __restrict__ W1t, const float* __restrict__ b1,
    const _Float16* __restrict__ W2t, const float* __restrict__ b2,
    float* __restrict__ ms, float* __restrict__ mv, int vmul) {
  __shared__ __attribute__((aligned(16))) _Float16 hT[4][16 * 128];
  const int lane = threadIdx.x & 31, w = threadIdx.x >> 5;
  const int row = lane & 15, half = lane >> 4;
  const int e0 = (blockIdx.x * 4 + w) * 16;
  const int eM = e0 + row;
  const int jM = ej[eM], iM = ei[eM];
  const _Float16* si = sf16 + (size_t)iM * 128;
  const _Float16* sj = sf16 + (size_t)jM * 128;
  const _Float16* de = deh + (size_t)eM * 32;

  // ---- GEMM1: 16x288 @ 288x128 -> 16x128, K=288 = 9 k-steps
  v8f c1[8];
#pragma unroll
  for (int t = 0; t < 8; ++t)
#pragma unroll
    for (int q = 0; q < 8; ++q) c1[t][q] = 0.f;

#pragma unroll
  for (int ks = 0; ks < 9; ++ks) {
    int ka = ks * 32 + half * 16;
    v16h a = ldfrag(asrc(si, sj, de, ka), asrc(si, sj, de, ka + 8));
    v16h bf[8];
#pragma unroll
    for (int t = 0; t < 8; ++t) {
      const _Float16* wr = W1t + (size_t)(t * 16 + row) * 288 + ks * 32 + half * 16;
      bf[t] = ldfrag(wr, wr + 8);
    }
    SCHED_FENCE();
#pragma unroll
    for (int t = 0; t < 8; ++t) c1[t] = wmma_f16(a, bf[t], c1[t]);
    SCHED_FENCE();
  }
  // bias + SiLU -> LDS (per-wave tile, row-major [16][128])
#pragma unroll
  for (int t = 0; t < 8; ++t) {
    int n = t * 16 + row;
    float bb = b1[n];
#pragma unroll
    for (int v = 0; v < 8; ++v) {
      float x = c1[t][v] + bb;
      x = x * (1.f / (1.f + __expf(-x)));
      hT[w][(v + half * 8) * 128 + n] = (_Float16)x;
    }
  }
  __syncthreads();

  // ---- A-fragments for GEMM2 (K=128 -> 4 k-steps), kept resident
  v16h aa[4];
#pragma unroll
  for (int ks = 0; ks < 4; ++ks) {
    const _Float16* p = &hT[w][row * 128 + ks * 32 + half * 16];
    aa[ks] = ldfrag(p, p + 8);
  }

  // Per-lane caches of edge-row data used by the scatter epilogue
  // (edge e = e0 + v + half*8 for vgpr v); avoids 8x redundant reloads.
  int   idxI[8], idxJ[8];
  float cutv[8];
#pragma unroll
  for (int v = 0; v < 8; ++v) {
    int e = e0 + v + half * 8;
    idxI[v] = ei[e];
    idxJ[v] = ej[e];
    cutv[v] = cutE[e];
  }

  // ---- GEMM2 chunks: 0=aij, 1=vij0(r), 2=vij1(vb[j]), 3=vij2(cross)
  for (int cc = 0; cc <= vmul; ++cc) {
    v8f c2[8];
#pragma unroll
    for (int t = 0; t < 8; ++t)
#pragma unroll
      for (int q = 0; q < 8; ++q) c2[t][q] = 0.f;
#pragma unroll
    for (int ks = 0; ks < 4; ++ks) {
      // two groups of 4 B-fragments to bound register pressure (aa resident)
#pragma unroll
      for (int g = 0; g < 2; ++g) {
        v16h bf[4];
#pragma unroll
        for (int u = 0; u < 4; ++u) {
          int t = g * 4 + u;
          const _Float16* wr =
              W2t + (size_t)(cc * 128 + t * 16 + row) * 128 + ks * 32 + half * 16;
          bf[u] = ldfrag(wr, wr + 8);
        }
        SCHED_FENCE();
#pragma unroll
        for (int u = 0; u < 4; ++u)
          c2[g * 4 + u] = wmma_f16(aa[ks], bf[u], c2[g * 4 + u]);
        SCHED_FENCE();
      }
    }
#pragma unroll
    for (int t = 0; t < 8; ++t) {
      int n = t * 16 + row;
      float bb = b2[cc * 128 + n];
#pragma unroll
      for (int v = 0; v < 8; ++v) {
        int e = e0 + v + half * 8;
        float val = (c2[t][v] + bb) * cutv[v];
        int jj = idxJ[v], ii = idxI[v];
        if (cc == 0) {
          atomicAdd(&ms[(size_t)ii * 128 + n], val * sb[(size_t)jj * 128 + n]);
        } else if (cc == 1) {
#pragma unroll
          for (int dd = 0; dd < 3; ++dd)
            atomicAdd(&mv[((size_t)ii * 3 + dd) * 128 + n], val * rvec[(size_t)e * 3 + dd]);
        } else if (cc == 2) {
#pragma unroll
          for (int dd = 0; dd < 3; ++dd)
            atomicAdd(&mv[((size_t)ii * 3 + dd) * 128 + n],
                      val * vb[((size_t)jj * 3 + dd) * 128 + n]);
        } else {
          float ax = vcur[((size_t)ii * 3 + 0) * 128 + n];
          float ay = vcur[((size_t)ii * 3 + 1) * 128 + n];
          float az = vcur[((size_t)ii * 3 + 2) * 128 + n];
          float bx = vb[((size_t)jj * 3 + 0) * 128 + n];
          float by = vb[((size_t)jj * 3 + 1) * 128 + n];
          float bz = vb[((size_t)jj * 3 + 2) * 128 + n];
          atomicAdd(&mv[((size_t)ii * 3 + 0) * 128 + n], val * (ay * bz - az * by));
          atomicAdd(&mv[((size_t)ii * 3 + 1) * 128 + n], val * (az * bx - ax * bz));
          atomicAdd(&mv[((size_t)ii * 3 + 2) * 128 + n], val * (ax * by - ay * bx));
        }
      }
    }
  }
}

// ------------------------------------------------------------- node updates
__global__ void k_update(float* __restrict__ s, const float* __restrict__ ms,
                         float* __restrict__ v, const float* __restrict__ mv,
                         const float* __restrict__ cnt, int N) {
  int i = blockIdx.x * blockDim.x + threadIdx.x;
  if (i >= N * 128) return;
  int n = i >> 7, h = i & 127;
  s[i] += ms[i];
  float ic = 1.f / fmaxf(cnt[n], 1.f);
#pragma unroll
  for (int dd = 0; dd < 3; ++dd) {
    size_t o = ((size_t)n * 3 + dd) * 128 + h;
    v[o] += mv[o] * ic;
  }
}

__global__ void k_vncat(const float* __restrict__ vv, const float* __restrict__ s,
                        _Float16* __restrict__ cat, int N) {
  int i = blockIdx.x * blockDim.x + threadIdx.x;
  if (i >= N * 128) return;
  int n = i >> 7, h = i & 127;
  float acc = 0.f;
#pragma unroll
  for (int dd = 0; dd < 3; ++dd) {
    float x = vv[((size_t)n * 3 + dd) * 256 + h];
    acc += x * x;
  }
  cat[(size_t)n * 256 + h]       = (_Float16)s[i];
  cat[(size_t)n * 256 + 128 + h] = (_Float16)sqrtf(fmaxf(acc, 1e-6f));
}

__global__ void k_gate(const float* __restrict__ h2, const float* __restrict__ vv,
                       _Float16* __restrict__ gvv, float* __restrict__ s, int N) {
  int i = blockIdx.x * blockDim.x + threadIdx.x;
  if (i >= N * 128) return;
  int n = i >> 7, h = i & 127;
  float g = h2[(size_t)n * 256 + h];
  s[i] += h2[(size_t)n * 256 + 128 + h];
#pragma unroll
  for (int dd = 0; dd < 3; ++dd)
    gvv[((size_t)n * 3 + dd) * 128 + h] =
        (_Float16)(g * vv[((size_t)n * 3 + dd) * 256 + 128 + h]);
}

__global__ void k_pool(const float* __restrict__ s, const int* __restrict__ batch,
                       float* __restrict__ gsum, float* __restrict__ gcnt, int N) {
  int i = blockIdx.x * blockDim.x + threadIdx.x;
  if (i >= N * 128) return;
  int n = i >> 7, h = i & 127;
  int b = batch[n];
  atomicAdd(&gsum[(size_t)b * 128 + h], s[i]);
  if (h == 0) atomicAdd(&gcnt[b], 1.f);
}

__global__ void k_final(const float* __restrict__ gsum, const float* __restrict__ gcnt,
                        float* __restrict__ out) {
  int i = blockIdx.x * blockDim.x + threadIdx.x;
  if (i >= 32 * 128) return;
  out[i] = gsum[i] / fmaxf(gcnt[i >> 7], 1.f);
}

// ===========================================================================
extern "C" void kernel_launch(void* const* d_in, const int* in_sizes, int n_in,
                              void* d_out, int out_size, void* d_ws, size_t ws_size,
                              hipStream_t stream) {
  const int N = in_sizes[0];          // 16384
  const int E = in_sizes[3] / 2;      // 262144
  const int N3 = 3 * N;

  // ---- input leaves: setup_inputs() dict insertion order, recursive -------
  const int*   z     = (const int*)d_in[0];
  const float* pos   = (const float*)d_in[1];
  const int*   batch = (const int*)d_in[2];
  const int*   eidx  = (const int*)d_in[3];
  const int*   ej    = eidx;          // edge_index[0] = j (source)
  const int*   ei    = eidx + E;      // edge_index[1] = i (destination)
  const float* emb   = (const float*)d_in[4];
  const float* nemb  = (const float*)d_in[5];
  const float* dpW   = (const float*)d_in[6];
  const float* dpB   = (const float*)d_in[7];
  const float* cbW   = (const float*)d_in[8];
  const float* cbB   = (const float*)d_in[9];
  struct LayerIn {
    const float *e1W, *e1b, *e2W, *e2b, *scW, *scb, *Wv0, *Ws1W, *Ws1b, *Ws2W, *Ws2b, *Wv1, *vecW;
  } L[5];
  {
    int idx = 10;
    for (int l = 0; l < 5; ++l) {
      L[l].e1W  = (const float*)d_in[idx + 0];
      L[l].e1b  = (const float*)d_in[idx + 1];
      L[l].e2W  = (const float*)d_in[idx + 2];
      L[l].e2b  = (const float*)d_in[idx + 3];
      L[l].scW  = (const float*)d_in[idx + 4];
      L[l].scb  = (const float*)d_in[idx + 5];
      L[l].Wv0  = (const float*)d_in[idx + 6];
      L[l].Ws1W = (const float*)d_in[idx + 7];
      L[l].Ws1b = (const float*)d_in[idx + 8];
      L[l].Ws2W = (const float*)d_in[idx + 9];
      L[l].Ws2b = (const float*)d_in[idx + 10];
      L[l].Wv1  = (const float*)d_in[idx + 11];
      L[l].vecW = (l > 0) ? (const float*)d_in[idx + 12] : nullptr;
      idx += (l > 0) ? 13 : 12;
    }
  }

  // ---- workspace bump allocator (256B aligned) ----------------------------
  char* base = (char*)d_ws;
  size_t off = 0;
  auto alloc = [&](size_t bytes) -> void* {
    off = (off + 255) & ~(size_t)255;
    void* p = base + off;
    off += bytes;
    return p;
  };
  float*    dArr = (float*)alloc((size_t)E * 4);
  float*    cutE = (float*)alloc((size_t)E * 4);
  float*    rvec = (float*)alloc((size_t)E * 12);
  _Float16* deh  = (_Float16*)alloc((size_t)E * 32 * 2);
  float*    cnt  = (float*)alloc((size_t)N * 4);
  float*    s    = (float*)alloc((size_t)N * 128 * 4);
  _Float16* sf16 = (_Float16*)alloc((size_t)N * 128 * 2);
  float*    sb   = (float*)alloc((size_t)N * 128 * 4);
  float*    ms   = (float*)alloc((size_t)N * 128 * 4);
  float*    v    = (float*)alloc((size_t)N3 * 128 * 4);
  float*    vb   = (float*)alloc((size_t)N3 * 128 * 4);
  float*    mv   = (float*)alloc((size_t)N3 * 128 * 4);
  _Float16* vf16 = (_Float16*)alloc((size_t)N3 * 128 * 2);
  float*    vv   = (float*)alloc((size_t)N3 * 256 * 4);
  _Float16* cat  = (_Float16*)alloc((size_t)N * 256 * 2);
  float*    agg  = (float*)alloc((size_t)N * 128 * 4);
  _Float16* hf16 = (_Float16*)alloc((size_t)N * 128 * 2);
  float*    h2   = (float*)alloc((size_t)N * 256 * 4);
  _Float16* gvv  = (_Float16*)alloc((size_t)N3 * 128 * 2);
  float*    gsum = (float*)alloc(32 * 128 * 4);
  float*    gcnt = (float*)alloc(32 * 4);
  _Float16* combT = (_Float16*)alloc((size_t)128 * 256 * 2);
  struct LayerW { _Float16 *W1t, *W2t, *scT, *vecT, *Wv0t, *Ws1t, *Ws2t, *Wv1t; } LW[5];
  for (int l = 0; l < 5; ++l) {
    int vmul = (l > 0) ? 3 : 1;
    LW[l].W1t  = (_Float16*)alloc((size_t)128 * 288 * 2);
    LW[l].W2t  = (_Float16*)alloc((size_t)(vmul + 1) * 128 * 128 * 2);
    LW[l].scT  = (_Float16*)alloc((size_t)128 * 128 * 2);
    LW[l].vecT = (_Float16*)alloc((size_t)128 * 128 * 2);
    LW[l].Wv0t = (_Float16*)alloc((size_t)256 * 128 * 2);
    LW[l].Ws1t = (_Float16*)alloc((size_t)128 * 256 * 2);
    LW[l].Ws2t = (_Float16*)alloc((size_t)256 * 128 * 2);
    LW[l].Wv1t = (_Float16*)alloc((size_t)128 * 128 * 2);
  }
  (void)ws_size;

  auto nb = [](size_t n) { return (unsigned)((n + 255) / 256); };
  auto zero = [&](float* p, size_t n) { k_zero<<<nb(n), 256, 0, stream>>>(p, n); };
  auto trT = [&](const float* src, _Float16* dst, int K, int Nn) {
    k_transT<<<nb((size_t)K * Nn), 256, 0, stream>>>(src, dst, K, Nn);
  };
  auto gemm = [&](const _Float16* A, const _Float16* Wt, int K, const float* bias,
                  float* C, _Float16* Ch, int M, int NT, int flags) {
    dim3 g(M / 64, NT / 128);
    k_gemm<<<g, 128, 0, stream>>>(A, Wt, K, bias, C, Ch, NT, flags);
  };

  // ---- weight conversion (f32 [K,N] -> f16 [N,K]) -------------------------
  trT(cbW, combT, 256, 128);
  for (int l = 0; l < 5; ++l) {
    int vmul = (l > 0) ? 3 : 1;
    trT(L[l].e1W, LW[l].W1t, 288, 128);
    trT(L[l].e2W, LW[l].W2t, 128, (vmul + 1) * 128);
    trT(L[l].scW, LW[l].scT, 128, 128);
    if (l > 0) trT(L[l].vecW, LW[l].vecT, 128, 128);
    trT(L[l].Wv0, LW[l].Wv0t, 128, 256);
    trT(L[l].Ws1W, LW[l].Ws1t, 256, 128);
    trT(L[l].Ws2W, LW[l].Ws2t, 128, 256);
    trT(L[l].Wv1, LW[l].Wv1t, 128, 128);
  }

  // ---- geometry + neighbor embedding --------------------------------------
  zero(cnt, N);
  zero(agg, (size_t)N * 128);
  zero(v, (size_t)N3 * 128);
  zero(gsum, 32 * 128);
  zero(gcnt, 32);
  k_geom<<<E / 256, 256, 0, stream>>>(pos, ej, ei, dArr, cutE, rvec, deh, cnt, E);
  k_nbr<<<E / 2, 256, 0, stream>>>(dArr, cutE, z, ej, ei, dpW, dpB, nemb, agg);
  k_cat0<<<nb((size_t)N * 128), 256, 0, stream>>>(z, emb, agg, cat, N);
  gemm(cat, combT, 256, cbB, s, sf16, N, 128, 0);   // s (f32) + sf16 in one pass

  // ---- 5 EQGAT conv layers -------------------------------------------------
  for (int l = 0; l < 5; ++l) {
    int vmul = (l > 0) ? 3 : 1;
    gemm(sf16, LW[l].scT, 128, L[l].scb, sb, nullptr, N, 128, 0);      // scalar_net
    if (l > 0) {
      k_f2h<<<nb((size_t)N3 * 128), 256, 0, stream>>>(v, vf16, (size_t)N3 * 128);
      gemm(vf16, LW[l].vecT, 128, nullptr, vb, nullptr, N3, 128, 0);   // vector_net
    }
    zero(ms, (size_t)N * 128);
    zero(mv, (size_t)N3 * 128);
    k_edge_conv<<<E / 64, 128, 0, stream>>>(sf16, deh, cutE, rvec, sb, v,
                                            (l > 0) ? vb : v, ej, ei,
                                            LW[l].W1t, L[l].e1b, LW[l].W2t, L[l].e2b,
                                            ms, mv, vmul);
    k_update<<<nb((size_t)N * 128), 256, 0, stream>>>(s, ms, v, mv, cnt, N);
    // GatedEquivBlock
    k_f2h<<<nb((size_t)N3 * 128), 256, 0, stream>>>(v, vf16, (size_t)N3 * 128);
    gemm(vf16, LW[l].Wv0t, 128, nullptr, vv, nullptr, N3, 256, 0);     // Wv0
    k_vncat<<<nb((size_t)N * 128), 256, 0, stream>>>(vv, s, cat, N);
    gemm(cat, LW[l].Ws1t, 256, L[l].Ws1b, nullptr, hf16, N, 128, FLAG_SILU);
    gemm(hf16, LW[l].Ws2t, 128, L[l].Ws2b, h2, nullptr, N, 256, 0);
    k_gate<<<nb((size_t)N * 128), 256, 0, stream>>>(h2, vv, gvv, s, N);
    gemm(gvv, LW[l].Wv1t, 128, nullptr, v, nullptr, N3, 128, FLAG_ACCUM); // v += dv
    k_f2h<<<nb((size_t)N * 128), 256, 0, stream>>>(s, sf16, (size_t)N * 128);
  }

  // ---- graph mean pooling --------------------------------------------------
  k_pool<<<nb((size_t)N * 128), 256, 0, stream>>>(s, batch, gsum, gcnt, N);
  k_final<<<16, 256, 0, stream>>>(gsum, gcnt, (float*)d_out);
  (void)n_in; (void)out_size;
}
